// MambaBlock_89919435309510
// MI455X (gfx1250) — compile-verified, matrix operand order checked by reading, not run on hardware
//
#include <hip/hip_runtime.h>
#include <stdint.h>

// ---------------------------------------------------------------------------
// Types for CDNA5 WMMA (wave32): v16bf A/B fragments, v8f accumulator.
// ---------------------------------------------------------------------------
typedef __attribute__((ext_vector_type(16))) __bf16 v16bf;
typedef __attribute__((ext_vector_type(8)))  float  v8f;

union FragBF { uint4 q[2]; v16bf v; };

__device__ __forceinline__ unsigned short f2bf(float f) {
    unsigned u = __float_as_uint(f);
    unsigned r = u + 0x7FFFu + ((u >> 16) & 1u);   // round-to-nearest-even
    return (unsigned short)(r >> 16);
}

__device__ __forceinline__ float softplus_f(float x) {
    return (x > 20.f) ? x : log1pf(__expf(x));
}

// ---------------------------------------------------------------------------
// bf16 WMMA GEMM:  C[M][N] (f32) = A[M][K] (bf16, row-major) * Bt[N][K]^T
// Wave tile 64x32: 4 A-frags x 2 B-frags -> 8 accumulators (64 VGPRs).
// Per K=32 step: 12 b128 loads feed 8 v_wmma (1.5 loads/WMMA).
// 4 waves per block stacked along M (block tile 256x32). Requires M%256==0,
// N%32==0, K%32==0 — true for every GEMM in this Mamba block.
// Per the ISA 16-bit A layout, each lane's v16bf fragment is two contiguous
// 16-byte chunks: [k0 + hi*8, +8) and [k0 + 16 + hi*8, +8).
// EPI==1: out = softplus(acc + bias[col])  (fused delta projection epilogue)
// ---------------------------------------------------------------------------
template<int EPI>
__global__ __launch_bounds__(128)
void gemm_bf16_wmma(const unsigned short* __restrict__ A,
                    const unsigned short* __restrict__ Bt,
                    float* __restrict__ C,
                    const float* __restrict__ bias,
                    int M, int N, int K)
{
    const int lane   = threadIdx.x & 31;
    const int wave   = threadIdx.x >> 5;
    const int lane16 = lane & 15;
    const int hi     = lane >> 4;

    const int n0 = blockIdx.x * 32;
    const int m0 = blockIdx.y * 256 + wave * 64;

    const unsigned short* ap[4];
    #pragma unroll
    for (int i = 0; i < 4; ++i)
        ap[i] = A + (size_t)(m0 + 16 * i + lane16) * K;
    const unsigned short* bp[2];
    bp[0] = Bt + (size_t)(n0 + lane16)      * K;
    bp[1] = Bt + (size_t)(n0 + 16 + lane16) * K;

    v8f acc[4][2];
    #pragma unroll
    for (int i = 0; i < 4; ++i)
        #pragma unroll
        for (int j = 0; j < 2; ++j)
            acc[i][j] = (v8f){};

    const int koff = hi * 8;
    for (int k0 = 0; k0 < K; k0 += 32) {
        if (k0 + 128 < K) {
            __builtin_prefetch(ap[0] + k0 + 128, 0, 3);   // global_prefetch_b8, WGP scope
            __builtin_prefetch(bp[0] + k0 + 128, 0, 3);
        }
        FragBF a[4], b[2];
        #pragma unroll
        for (int i = 0; i < 4; ++i) {
            a[i].q[0] = *(const uint4*)(ap[i] + k0 + koff);
            a[i].q[1] = *(const uint4*)(ap[i] + k0 + 16 + koff);
        }
        #pragma unroll
        for (int j = 0; j < 2; ++j) {
            b[j].q[0] = *(const uint4*)(bp[j] + k0 + koff);
            b[j].q[1] = *(const uint4*)(bp[j] + k0 + 16 + koff);
        }
        #pragma unroll
        for (int i = 0; i < 4; ++i)
            #pragma unroll
            for (int j = 0; j < 2; ++j)
                acc[i][j] = __builtin_amdgcn_wmma_f32_16x16x32_bf16(
                    false, a[i].v, false, b[j].v, (short)0, acc[i][j], false, false);
    }

    // C/D layout: VGPR r -> row (r + hi*8), col lane16 within 16x16 tile.
    const int colA = n0 + lane16;
    const int colB = n0 + 16 + lane16;
    float biasA = (EPI == 1) ? bias[colA] : 0.f;
    float biasB = (EPI == 1) ? bias[colB] : 0.f;
    #pragma unroll
    for (int i = 0; i < 4; ++i) {
        #pragma unroll
        for (int r = 0; r < 8; ++r) {
            int row = m0 + 16 * i + hi * 8 + r;
            float v0 = acc[i][0][r], v1 = acc[i][1][r];
            if (EPI == 1) {
                v0 = softplus_f(v0 + biasA);
                v1 = softplus_f(v1 + biasB);
            }
            C[(size_t)row * N + colA] = v0;
            C[(size_t)row * N + colB] = v1;
        }
    }
}

// ---------------------------------------------------------------------------
// f32 -> bf16 convert (same layout)
// ---------------------------------------------------------------------------
__global__ void convert_bf16(const float* __restrict__ in,
                             unsigned short* __restrict__ out, int n)
{
    int i = blockIdx.x * blockDim.x + threadIdx.x;
    if (i < n) out[i] = f2bf(in[i]);
}

// ---------------------------------------------------------------------------
// Transpose f32 [K][N] -> bf16 [N][K] via LDS 32x33 tile (conflict-free).
// ---------------------------------------------------------------------------
__global__ __launch_bounds__(256)
void transpose_f32_to_bf16(const float* __restrict__ in,
                           unsigned short* __restrict__ out, int K, int N)
{
    __shared__ float tile[32][33];
    const int k0 = blockIdx.y * 32, n0 = blockIdx.x * 32;
    const int tx = threadIdx.x & 31, ty = threadIdx.x >> 5;
    for (int i = ty; i < 32; i += 8) {
        int k = k0 + i, n = n0 + tx;
        if (k < K && n < N) tile[i][tx] = in[(size_t)k * N + n];
    }
    __syncthreads();
    for (int i = ty; i < 32; i += 8) {
        int n = n0 + i, k = k0 + tx;
        if (n < N && k < K) out[(size_t)n * K + k] = f2bf(tile[tx][i]);
    }
}

// ---------------------------------------------------------------------------
// Causal depthwise conv (K=3) + SiLU. xr is [L][4096]; channel d = col d.
// Writes f32 (for the scan's u) and bf16 (A-matrix of the W_x GEMM).
// ---------------------------------------------------------------------------
__global__ __launch_bounds__(256)
void conv_silu_kernel(const float* __restrict__ xr, const float* __restrict__ cw,
                      float* __restrict__ xc, unsigned short* __restrict__ xcb)
{
    int idx = blockIdx.x * blockDim.x + threadIdx.x;   // L*DIN = 2048*2048
    int l = idx >> 11;
    int d = idx & 2047;
    float w0 = cw[d * 3 + 0], w1 = cw[d * 3 + 1], w2 = cw[d * 3 + 2];
    float acc = w2 * xr[(size_t)l * 4096 + d];
    if (l >= 1) acc += w1 * xr[(size_t)(l - 1) * 4096 + d];
    if (l >= 2) acc += w0 * xr[(size_t)(l - 2) * 4096 + d];
    float s = acc / (1.f + __expf(-acc));              // SiLU
    xc[idx]  = s;
    xcb[idx] = f2bf(s);
}

// ---------------------------------------------------------------------------
// Pack delta_lr (cols 0..63 of x_dbl[L][96]) into bf16 [L][64].
// ---------------------------------------------------------------------------
__global__ void pack_dlr(const float* __restrict__ xdbl,
                         unsigned short* __restrict__ dlrb)
{
    int idx = blockIdx.x * blockDim.x + threadIdx.x;   // L*64
    int l = idx >> 6, j = idx & 63;
    dlrb[idx] = f2bf(xdbl[(size_t)l * 96 + j]);
}

// ---------------------------------------------------------------------------
// Selective scan. The reference's flip/cumsum formulation equals the
// recurrence h[l] = exp(delta*A)*h[l-1] + delta*u*B[l];  y = <h, C[l]>.
// One thread per (d, n): 32768 threads = 1024 wave32. The 16-lane sum over n
// is a shfl_xor tree (n occupies lane bits [3:0], so xor 8/4/2/1 stays in
// group). Epilogue fuses +u*D and *silu(res), emitting bf16 for the W_out
// GEMM's A matrix.
// ---------------------------------------------------------------------------
__global__ __launch_bounds__(256)
void scan_kernel(const float* __restrict__ delta, const float* __restrict__ u,
                 const float* __restrict__ xdbl,  const float* __restrict__ A_log,
                 const float* __restrict__ Dp,    const float* __restrict__ xr,
                 unsigned short* __restrict__ yb)
{
    int t = blockIdx.x * blockDim.x + threadIdx.x;     // 0..32767
    int d = t >> 4;
    int n = t & 15;
    float An = -__expf(A_log[d * 16 + n]);
    float Dd = Dp[d];
    float h = 0.f;
    for (int l = 0; l < 2048; ++l) {
        float dt = delta[(size_t)l * 2048 + d];
        float uv = u[(size_t)l * 2048 + d];
        float Bn = xdbl[(size_t)l * 96 + 64 + n];
        float Cn = xdbl[(size_t)l * 96 + 80 + n];
        float w = __expf(dt * An);
        h = h * w + dt * uv * Bn;
        float part = h * Cn;
        part += __shfl_xor(part, 8, 32);
        part += __shfl_xor(part, 4, 32);
        part += __shfl_xor(part, 2, 32);
        part += __shfl_xor(part, 1, 32);
        if (n == 0) {
            float y = part + uv * Dd;
            float r = xr[(size_t)l * 4096 + 2048 + d]; // res half
            float o = y * (r / (1.f + __expf(-r)));    // * silu(res)
            yb[(size_t)l * 2048 + d] = f2bf(o);
        }
    }
}

// ---------------------------------------------------------------------------
// Launch: conversions -> GEMM1 -> conv/SiLU -> GEMM2 -> pack -> delta-GEMM
// (softplus fused) -> scan -> GEMM3 (output).
// ---------------------------------------------------------------------------
extern "C" void kernel_launch(void* const* d_in, const int* in_sizes, int n_in,
                              void* d_out, int out_size, void* d_ws, size_t ws_size,
                              hipStream_t stream)
{
    (void)in_sizes; (void)n_in; (void)out_size; (void)ws_size;
    const float* x       = (const float*)d_in[0];
    const float* W_in    = (const float*)d_in[1];
    const float* conv_w  = (const float*)d_in[2];
    const float* W_x     = (const float*)d_in[3];
    const float* W_delta = (const float*)d_in[4];
    const float* b_delta = (const float*)d_in[5];
    const float* A_log   = (const float*)d_in[6];
    const float* Dp      = (const float*)d_in[7];
    const float* W_out   = (const float*)d_in[8];
    float* out = (float*)d_out;

    const int SEQ = 2048, DM = 1024, DIN = 2048, DTR = 64;

    char* p = (char*)d_ws;
    auto carve = [&](size_t bytes) -> char* {
        char* r = p; p += (bytes + 255) & ~(size_t)255; return r;
    };
    unsigned short* xb   = (unsigned short*)carve(sizeof(short) * SEQ * DM);       // x bf16 [2048][1024]
    unsigned short* WinT = (unsigned short*)carve(sizeof(short) * 2 * DIN * DM);   // [4096][1024]
    unsigned short* WxT  = (unsigned short*)carve(sizeof(short) * 96 * DIN);       // [96][2048]
    unsigned short* WdT  = (unsigned short*)carve(sizeof(short) * DIN * DTR);      // [2048][64]
    unsigned short* WoT  = (unsigned short*)carve(sizeof(short) * DM * DIN);       // [1024][2048]
    float*          xr   = (float*)carve(sizeof(float) * SEQ * 2 * DIN);           // x@W_in [2048][4096]
    float*          xc   = (float*)carve(sizeof(float) * SEQ * DIN);               // conv+silu f32
    unsigned short* xcb  = (unsigned short*)carve(sizeof(short) * SEQ * DIN);      // conv+silu bf16
    float*          xdbl = (float*)carve(sizeof(float) * SEQ * 96);                // [2048][96]
    unsigned short* dlrb = (unsigned short*)carve(sizeof(short) * SEQ * DTR);      // delta_lr bf16
    float*          delta= (float*)carve(sizeof(float) * SEQ * DIN);               // softplus(delta)
    unsigned short* yb   = (unsigned short*)carve(sizeof(short) * SEQ * DIN);      // scan out bf16

    // Stage operands as bf16 (weights transposed to [N][K]).
    convert_bf16<<<(SEQ * DM + 255) / 256, 256, 0, stream>>>(x, xb, SEQ * DM);
    transpose_f32_to_bf16<<<dim3((2 * DIN + 31) / 32, (DM + 31) / 32), 256, 0, stream>>>(W_in, WinT, DM, 2 * DIN);
    transpose_f32_to_bf16<<<dim3((96 + 31) / 32, (DIN + 31) / 32), 256, 0, stream>>>(W_x, WxT, DIN, 96);
    transpose_f32_to_bf16<<<dim3((DIN + 31) / 32, (DTR + 31) / 32), 256, 0, stream>>>(W_delta, WdT, DTR, DIN);
    transpose_f32_to_bf16<<<dim3((DM + 31) / 32, (DIN + 31) / 32), 256, 0, stream>>>(W_out, WoT, DIN, DM);

    // GEMM1: x_and_res = x @ W_in   (2048x4096, K=1024)
    gemm_bf16_wmma<0><<<dim3((2 * DIN) / 32, SEQ / 256), 128, 0, stream>>>(xb, WinT, xr, nullptr, SEQ, 2 * DIN, DM);

    // Depthwise causal conv + SiLU on xi half.
    conv_silu_kernel<<<(SEQ * DIN) / 256, 256, 0, stream>>>(xr, conv_w, xc, xcb);

    // GEMM2: x_dbl = xi @ W_x   (2048x96, K=2048)
    gemm_bf16_wmma<0><<<dim3(96 / 32, SEQ / 256), 128, 0, stream>>>(xcb, WxT, xdbl, nullptr, SEQ, 96, DIN);

    pack_dlr<<<(SEQ * DTR) / 256, 256, 0, stream>>>(xdbl, dlrb);

    // delta = softplus(delta_lr @ W_delta + b_delta)   (2048x2048, K=64, fused epilogue)
    gemm_bf16_wmma<1><<<dim3(DIN / 32, SEQ / 256), 128, 0, stream>>>(dlrb, WdT, delta, b_delta, SEQ, DIN, DTR);

    // Selective scan + *silu(res) epilogue -> bf16 y.
    scan_kernel<<<(DIN * 16) / 256, 256, 0, stream>>>(delta, xc, xdbl, A_log, Dp, xr, yb);

    // GEMM3: out = y @ W_out   (2048x1024, K=2048)
    gemm_bf16_wmma<0><<<dim3(DM / 32, SEQ / 256), 128, 0, stream>>>(yb, WoT, out, nullptr, SEQ, DM, DIN);
}
